// MDRNN_ver2_68367289417805
// MI455X (gfx1250) — matrix operand order repaired and checked by reading, not compiled, not run on hardware
//
#include <hip/hip_runtime.h>
#include <math.h>

// ---------------------------------------------------------------------------
// MDRNN (2-D bi-LSTM grid) for MI455X / gfx1250.
// B=64, T=32, P=64, I=256, H=128, O=88.  Scan of 2048 sequential steps.
// Strategy: single persistent 1024-thread workgroup (32 wave32) on one WGP,
// bf16 WMMA (v_wmma_f32_16x16x32_bf16) for all GEMMs, weights pre-packed into
// CDNA5 B-fragment layout in workspace (bf16, ~1.6MB -> L2 resident),
// h-state kept in A-fragment layout in 224KB dynamic LDS, c-state in VGPRs.
// Loop-invariant weight/bias addresses are offset by an opaque zero (volatile
// asm) INSIDE the scan loop: defeats LICM hoisting (no scratch spills) while
// preserving global address-space inference (global_load, not flat_load, so
// the weight stream tracks LOADcnt only and does not inflate DScnt).
// ---------------------------------------------------------------------------

typedef __attribute__((ext_vector_type(16))) __bf16 bf16x16;
typedef __attribute__((ext_vector_type(8)))  float  f32x8;

#define Bsz   64
#define Tn    32
#define Pn    64
#define In    256
#define Hn    128
#define On    88
#define NSTEP (Tn * Pn)          // 2048
#define G4H   (4 * Hn)           // 512

// workspace layout (ushort units for bf16 fragment regions)
#define OFF_WTIH  0              // 2 * 512*256            = 262144
#define OFF_WTHH  262144         // 2 * 512*128            = 131072
#define OFF_WPIH  393216         // 2 * 512*256            = 262144
#define OFF_WPHH  655360         // 2 * 512*128            = 131072
#define OFF_WO    786432         // 96 * 256               =  24576
#define WS_USHORTS 811008
#define OFF_BIAS_BYTES (WS_USHORTS * 2)   // then 1024 f32 biasT, 1024 f32 biasP

// LDS partition (ushort units)
#define LDS_XA    0              // x_t A-frags: 4 mt * 8 kt * 32 * 16 = 16384
#define LDS_TA    16384          // tout A-frags (h_tf kt0-3 | h_tb kt4-7) 16384
#define LDS_HPA   32768          // pitch h A-frags: 2 dirs * 4mt*4kt*32*16 = 16384
#define LDS_GS    49152          // gates f32 [64][512] = 32768 f32 (131072 B)
#define LDS_BYTES (49152 * 2 + 64 * 512 * 4)   // 229376 B < 320KB WGP LDS

// Opaque zero offset defined inside the loop body: address becomes formally
// loop-variant (kills LICM hoisting of the loads) but the base pointer's
// global address space stays visible => global_load_*, LOADcnt-only.
#define PIN_OFFSET(p)                                \
  {                                                  \
    unsigned _zero = 0;                              \
    asm volatile("" : "+s"(_zero));                  \
    p += _zero;                                      \
  }

static __device__ __forceinline__ unsigned short f2bf(float f) {
  unsigned u = __float_as_uint(f);
  unsigned r = (u + 0x7FFFu + ((u >> 16) & 1u)) >> 16;   // round-nearest-even
  return (unsigned short)r;
}
static __device__ __forceinline__ float sigmoidf(float v) {
  return 1.0f / (1.0f + __expf(-v));
}

// ---------------------------------------------------------------------------
// Phase 0: repack W (Nsrc x Kdim, row-major; used as B = W^T, K x N) into
// CDNA5 bf16 B-fragments.  32x16 tile layout:
//   lanes 0-15 : N = nt*16+lane,      K = kt*32 + (slot<8 ? slot : 8+slot)
//   lanes 16-31: N = nt*16+lane-16,   K = kt*32 + 8 + (slot<8 ? slot : 8+slot)
// frag element index = ((nt*nkt + kt)*32 + lane)*16 + slot
// ---------------------------------------------------------------------------
__global__ void build_bfrag(const float* __restrict__ W, unsigned short* __restrict__ dst,
                            int Kdim, int Ntot, int Nsrc) {
  int idx = blockIdx.x * blockDim.x + threadIdx.x;
  int total = Ntot * Kdim;
  if (idx >= total) return;
  int slot = idx & 15;
  int lane = (idx >> 4) & 31;
  int tile = idx >> 9;
  int nkt  = Kdim >> 5;
  int nt = tile / nkt, kt = tile - nt * nkt;
  int n = nt * 16 + (lane & 15);
  int k = kt * 32 + (slot & 7) + ((slot & 8) << 1) + ((lane >> 4) << 3);
  float v = (n < Nsrc) ? W[n * Kdim + k] : 0.0f;
  dst[idx] = f2bf(v);
}

__global__ void build_bias(const float* __restrict__ btih, const float* __restrict__ bthh,
                           const float* __restrict__ bpih, const float* __restrict__ bphh,
                           float* __restrict__ bT, float* __restrict__ bP) {
  int i = blockIdx.x * blockDim.x + threadIdx.x;
  if (i < 2 * G4H) {
    bT[i] = btih[i] + bthh[i];
    bP[i] = bpih[i] + bphh[i];
  }
}

// ---------------------------------------------------------------------------
// GEMM stage: gates[64x512] = A1[64 x 32*nk1] @ B1 + A2[64 x 32*nk2] @ B2
// 32 waves, wave w owns N-tile w (one 16-col stripe), loops M-tiles 0..3.
// B fragments are loaded once per call (L2 hit) and reused across 4 M-tiles.
// ---------------------------------------------------------------------------
static __device__ __forceinline__ void gemm_cell(
    int wave, int lane,
    const unsigned short* A1, int nktA1, int kt1off, int nk1,
    const unsigned short* __restrict__ B1,
    const unsigned short* A2, int nktA2, int kt2off, int nk2,
    const unsigned short* __restrict__ B2,
    float* gs) {
  PIN_OFFSET(B1);              // keep weight streaming inside the scan loop
  PIN_OFFSET(B2);
  int nt = wave;
  int ncol = (nt << 4) + (lane & 15);
  int mhi  = (lane >> 4) << 3;
#pragma unroll
  for (int mt = 0; mt < 4; ++mt) {
    f32x8 acc = {};
    for (int kt = 0; kt < nk1; ++kt) {
      bf16x16 a = *(const bf16x16*)(A1 + (((mt * nktA1) + kt1off + kt) * 32 + lane) * 16);
      bf16x16 b = *(const bf16x16*)(B1 + (((nt * nk1) + kt) * 32 + lane) * 16);
      acc = __builtin_amdgcn_wmma_f32_16x16x32_bf16(false, a, false, b, (short)0, acc, false, false);
    }
    for (int kt = 0; kt < nk2; ++kt) {
      bf16x16 a = *(const bf16x16*)(A2 + (((mt * nktA2) + kt2off + kt) * 32 + lane) * 16);
      bf16x16 b = *(const bf16x16*)(B2 + (((nt * nk2) + kt) * 32 + lane) * 16);
      acc = __builtin_amdgcn_wmma_f32_16x16x32_bf16(false, a, false, b, (short)0, acc, false, false);
    }
    int mrow = (mt << 4) + mhi;
#pragma unroll
    for (int r = 0; r < 8; ++r)           // D-frag: m = r + 8*(lane>>4), n = lane&15
      gs[(mrow + r) * G4H + ncol] = acc[r];
  }
}

// ---------------------------------------------------------------------------
// LSTM pointwise: c' = sig(f)*c + sig(i)*tanh(g); h' = sig(o)*tanh(c').
// Each thread owns 8 (b,j) elements; writes h' straight into A-frag layout.
// ---------------------------------------------------------------------------
static __device__ __forceinline__ void cell_update(
    int tid, const float* gs, const float* __restrict__ bias,
    float* c, unsigned short* Adst, int nktA, int ktoff) {
  PIN_OFFSET(bias);            // prevent hoisting 32 bias loads out of scan loop
  int b  = tid >> 4;
  int j0 = (tid & 15) << 3;
  const float* row = gs + b * G4H;
#pragma unroll
  for (int r = 0; r < 8; ++r) {
    int j = j0 + r;
    float gi = row[j]           + bias[j];
    float gf = row[Hn + j]      + bias[Hn + j];
    float gg = row[2 * Hn + j]  + bias[2 * Hn + j];
    float go = row[3 * Hn + j]  + bias[3 * Hn + j];
    float cn = sigmoidf(gf) * c[r] + sigmoidf(gi) * tanhf(gg);
    float hn = sigmoidf(go) * tanhf(cn);
    c[r] = cn;
    // (m,k) -> (lane,slot) of 16x32 bf16 A-frag
    int mt = b >> 4, m = b & 15;
    int kk = j & 31, kt = (j >> 5) + ktoff;
    int laneHigh = (kk >> 3) & 1;
    int slot = (kk & 7) + (((kk >> 4) & 1) << 3);
    int lane = m + (laneHigh << 4);
    Adst[((mt * nktA + kt) * 32 + lane) * 16 + slot] = f2bf(hn);
  }
}

// ---------------------------------------------------------------------------
// Persistent scan kernel: one workgroup, 32 waves, 2048 sequential steps.
// ---------------------------------------------------------------------------
__global__ void __launch_bounds__(1024, 1)
mdrnn_scan(const float* __restrict__ xg,
           const unsigned short* __restrict__ WtihB, const unsigned short* __restrict__ WthhB,
           const unsigned short* __restrict__ WpihB, const unsigned short* __restrict__ WphhB,
           const unsigned short* __restrict__ WoB,
           const float* __restrict__ biasT, const float* __restrict__ biasP,
           const float* __restrict__ bo, float* __restrict__ out) {
  extern __shared__ unsigned short smem[];
  unsigned short* xA  = smem + LDS_XA;
  unsigned short* tA  = smem + LDS_TA;
  unsigned short* hpA = smem + LDS_HPA;
  float* gs = (float*)(smem + LDS_GS);

  const int tid  = threadIdx.x;
  const int wave = tid >> 5;
  const int lane = tid & 31;

  float ctf[8] = {}, ctb[8] = {}, cpf[8] = {}, cpb[8] = {};

  // zero h-state fragment buffers
  for (int k = tid; k < 16384; k += 1024) { tA[k] = 0; hpA[k] = 0; }
  __syncthreads();

  for (int s = 0; s < NSTEP; ++s) {
    // ---- load x_t[64x256] (f32 -> bf16 A-fragments) -----------------------
    const float* xs = xg;
    PIN_OFFSET(xs);
#pragma unroll 4
    for (int r = 0; r < 16; ++r) {
      int e = (tid << 4) + r;
      int b = e >> 8, i = e & 255;
      const float* src = xs + ((size_t)(b * NSTEP + s)) * In + i;
      float v = *src;
      if (s + 1 < NSTEP) __builtin_prefetch(src + In, 0, 1);  // global_prefetch_b8
      int mt = b >> 4, m = b & 15;
      int kk = i & 31, kt = i >> 5;
      int laneHigh = (kk >> 3) & 1;
      int slot = (kk & 7) + (((kk >> 4) & 1) << 3);
      int ln = m + (laneHigh << 4);
      xA[((mt * 8 + kt) * 32 + ln) * 16 + slot] = f2bf(v);
    }
    __syncthreads();

    // ---- time fwd: gates = x@Wt_ih[0]^T + h_tf@Wt_hh[0]^T -----------------
    gemm_cell(wave, lane, xA, 8, 0, 8, WtihB, tA, 8, 0, 4, WthhB, gs);
    __syncthreads();
    cell_update(tid, gs, biasT, ctf, tA, 8, 0);
    __syncthreads();

    // ---- time bwd ---------------------------------------------------------
    gemm_cell(wave, lane, xA, 8, 0, 8, WtihB + 131072, tA, 8, 4, 4, WthhB + 65536, gs);
    __syncthreads();
    cell_update(tid, gs, biasT + G4H, ctb, tA, 8, 4);
    __syncthreads();

    // ---- pitch fwd: gates = tout@Wp_ih[0]^T + h_pf@Wp_hh[0]^T -------------
    gemm_cell(wave, lane, tA, 8, 0, 8, WpihB, hpA, 4, 0, 4, WphhB, gs);
    __syncthreads();
    cell_update(tid, gs, biasP, cpf, hpA, 4, 0);
    __syncthreads();

    // ---- pitch bwd --------------------------------------------------------
    gemm_cell(wave, lane, tA, 8, 0, 8, WpihB + 131072, hpA + 8192, 4, 0, 4, WphhB + 65536, gs);
    __syncthreads();
    cell_update(tid, gs, biasP + G4H, cpb, hpA + 8192, 4, 0);
    __syncthreads();
  }

  // ---- out[64x88] = concat(h_pf,h_pb)[64x256] @ Wo^T + bo (N padded to 96)
  if (wave < 24) {
    int mt = wave / 6, nt = wave - mt * 6;
    f32x8 acc = {};
#pragma unroll
    for (int kt = 0; kt < 8; ++kt) {
      const unsigned short* Asrc = (kt < 4)
          ? hpA +        ((mt * 4 + kt)       * 32 + lane) * 16
          : hpA + 8192 + ((mt * 4 + (kt - 4)) * 32 + lane) * 16;
      bf16x16 a = *(const bf16x16*)Asrc;
      bf16x16 b = *(const bf16x16*)(WoB + ((nt * 8 + kt) * 32 + lane) * 16);
      acc = __builtin_amdgcn_wmma_f32_16x16x32_bf16(false, a, false, b, (short)0, acc, false, false);
    }
    int mhi = (lane >> 4) << 3;
    int n   = lane & 15;
#pragma unroll
    for (int r = 0; r < 8; ++r) {
      int brow = mt * 16 + mhi + r;
      int col  = nt * 16 + n;
      if (col < On) out[brow * On + col] = acc[r] + bo[col];
    }
  }
}

// ---------------------------------------------------------------------------
extern "C" void kernel_launch(void* const* d_in, const int* in_sizes, int n_in,
                              void* d_out, int out_size, void* d_ws, size_t ws_size,
                              hipStream_t stream) {
  (void)in_sizes; (void)n_in; (void)out_size; (void)ws_size;
  const float* x     = (const float*)d_in[0];
  const float* Wt_ih = (const float*)d_in[1];   // [2,512,256]
  const float* Wt_hh = (const float*)d_in[2];   // [2,512,128]
  const float* bt_ih = (const float*)d_in[3];
  const float* bt_hh = (const float*)d_in[4];
  const float* Wp_ih = (const float*)d_in[5];   // [2,512,256]
  const float* Wp_hh = (const float*)d_in[6];   // [2,512,128]
  const float* bp_ih = (const float*)d_in[7];
  const float* bp_hh = (const float*)d_in[8];
  const float* Wo    = (const float*)d_in[9];   // [88,256]
  const float* bo    = (const float*)d_in[10];
  float* out = (float*)d_out;

  unsigned short* wsu = (unsigned short*)d_ws;
  float* biasT = (float*)((char*)d_ws + OFF_BIAS_BYTES);
  float* biasP = biasT + 2 * G4H;

  // Phase 0: weight repack (f32 -> bf16 B-fragments) + bias fusion
  for (int d = 0; d < 2; ++d) {
    build_bfrag<<<512, 256, 0, stream>>>(Wt_ih + d * 131072, wsu + OFF_WTIH + d * 131072, 256, 512, 512);
    build_bfrag<<<256, 256, 0, stream>>>(Wt_hh + d *  65536, wsu + OFF_WTHH + d *  65536, 128, 512, 512);
    build_bfrag<<<512, 256, 0, stream>>>(Wp_ih + d * 131072, wsu + OFF_WPIH + d * 131072, 256, 512, 512);
    build_bfrag<<<256, 256, 0, stream>>>(Wp_hh + d *  65536, wsu + OFF_WPHH + d *  65536, 128, 512, 512);
  }
  build_bfrag<<<96, 256, 0, stream>>>(Wo, wsu + OFF_WO, 256, 96, 88);
  build_bias<<<4, 256, 0, stream>>>(bt_ih, bt_hh, bp_ih, bp_hh, biasT, biasP);

  // Phase 1: persistent sequential scan (one WGP, 32 waves, 224KB LDS)
  mdrnn_scan<<<1, 1024, LDS_BYTES, stream>>>(
      x,
      wsu + OFF_WTIH, wsu + OFF_WTHH, wsu + OFF_WPIH, wsu + OFF_WPHH, wsu + OFF_WO,
      biasT, biasP, bo, out);
}
